// TNEP_10445360464282
// MI455X (gfx1250) — matrix operand order repaired and checked by reading, not compiled, not run on hardware
//
#include <hip/hip_runtime.h>
#include <hip/hip_bf16.h>

typedef __attribute__((ext_vector_type(16))) _Float16 v16h;
typedef __attribute__((ext_vector_type(8)))  float    v8f;

#define NTYPES 8
#define DIMQ   64
#define NNEU   64

// CDNA5 hardware tanh (V_TANH_F32) when the toolchain exposes it.
static __device__ __forceinline__ float fast_tanh(float x) {
#if defined(__has_builtin)
#if __has_builtin(__builtin_amdgcn_tanhf)
    return __builtin_amdgcn_tanhf(x);
#else
    return tanhf(x);
#endif
#else
    return tanhf(x);
#endif
}

// ---------------- sort / prep kernels ----------------

__global__ void tnep_init(int* __restrict__ sortedIdx, int* __restrict__ counts,
                          int* __restrict__ cursors, int maxPad) {
    int i = blockIdx.x * blockDim.x + threadIdx.x;
    if (i < maxPad) sortedIdx[i] = -1;
    if (i < NTYPES) { counts[i] = 0; cursors[i] = 0; }
}

__global__ void tnep_hist(const int* __restrict__ Z, int* __restrict__ counts, int n) {
    int i = blockIdx.x * blockDim.x + threadIdx.x;
    if (i < n) atomicAdd(&counts[Z[i]], 1);
}

// 1 block: serial 8-wide padded exclusive scan, then parallel tile-type fill.
__global__ void tnep_setup(const int* __restrict__ counts, int* __restrict__ offsets,
                           int* __restrict__ tileType, int* __restrict__ tileCount) {
    __shared__ int soff[NTYPES + 1];
    if (threadIdx.x == 0) {
        int o = 0;
        for (int t = 0; t < NTYPES; ++t) {
            offsets[t] = o; soff[t] = o;
            o += ((counts[t] + 15) >> 4) << 4;     // pad each segment to 16 rows
        }
        offsets[NTYPES] = o; soff[NTYPES] = o;
        *tileCount = o >> 4;
    }
    __syncthreads();
    int nT = soff[NTYPES] >> 4;
    for (int t = threadIdx.x; t < nT; t += blockDim.x) {
        int row = t << 4;
        int ty = 0;
        while (ty < NTYPES - 1 && row >= soff[ty + 1]) ++ty;
        tileType[t] = ty;
    }
}

__global__ void tnep_scatter(const int* __restrict__ Z, const int* __restrict__ offsets,
                             int* __restrict__ cursors, int* __restrict__ sortedIdx, int n) {
    int i = blockIdx.x * blockDim.x + threadIdx.x;
    if (i < n) {
        int t = Z[i];
        int p = offsets[t] + atomicAdd(&cursors[t], 1);
        sortedIdx[p] = i;
    }
}

// W0[t][k][n] f32  ->  W0T[t][n][k] f16  (WMMA B-matrix friendly: contiguous K per column)
__global__ void tnep_wprep(const float* __restrict__ W0, _Float16* __restrict__ W0T) {
    int i = blockIdx.x * blockDim.x + threadIdx.x;
    if (i >= NTYPES * DIMQ * NNEU) return;
    int t = i >> 12, n = (i >> 6) & 63, k = i & 63;
    W0T[(t << 12) + (n << 6) + k] = (_Float16)W0[(t << 12) + (k << 6) + n];
}

// ---------------- WMMA grouped-GEMM + fused epilogue ----------------
// one wave32 per 16-atom (type-uniform) tile; 8x v_wmma_f32_16x16x32_f16 per wave
__global__ __launch_bounds__(256) void tnep_gemm(
    const float* __restrict__ desc, const float* __restrict__ b0,
    const float* __restrict__ W1, const _Float16* __restrict__ W0T,
    const int* __restrict__ sortedIdx, const int* __restrict__ tileType,
    const int* __restrict__ tileCountPtr, float* __restrict__ partials)
{
    const int lane = threadIdx.x & 31;
    const int tile = (blockIdx.x * blockDim.x + threadIdx.x) >> 5;   // wave-uniform
    const int nTiles = *tileCountPtr;
    if (tile >= nTiles) return;

    const int base    = tile << 4;
    const int type    = tileType[tile];
    const int halfsel = lane >> 4;      // 0: lanes 0-15, 1: lanes 16-31
    const int l15     = lane & 15;

    // ---- A fragments: 16x64 descriptors as two 16x32 f16 tiles ----
    // A layout: lane owns row M=lane&15; contiguous K-run [koff..koff+16), koff = 16*halfsel
    const int atomA = sortedIdx[base + l15];
    const int koff  = 16 * halfsel;
    v16h a[2];
    if (atomA >= 0) {
        const float* row = desc + (long)atomA * DIMQ;
#pragma unroll
        for (int kk = 0; kk < 2; ++kk) {
            const int k0 = 32 * kk + koff;
            const int k1 = 32 * kk + 8 + koff;
#pragma unroll
            for (int j = 0; j < 8; ++j) {
                a[kk][j]     = (_Float16)row[k0 + j];
                a[kk][8 + j] = (_Float16)row[k1 + j];
            }
        }
    } else {
#pragma unroll
        for (int kk = 0; kk < 2; ++kk)
#pragma unroll
            for (int j = 0; j < 16; ++j) a[kk][j] = (_Float16)0.0f;
    }

    // row-validity mask for this lane's 8 accumulator rows (M = rbase + r)
    const int rbase = 8 * halfsel;
    float mask[8];
#pragma unroll
    for (int r = 0; r < 8; ++r)
        mask[r] = (sortedIdx[base + rbase + r] >= 0) ? 1.0f : 0.0f;

    const _Float16* w0t = W0T + (type << 12);      // [n][k]
    float e = 0.0f;

#pragma unroll
    for (int nt = 0; nt < 4; ++nt) {
        const int n = nt * 16 + l15;
        // B layout: lane owns column n; lanes 0-15 K 0..15, lanes 16-31 K 16..31 (contiguous)
        const _Float16* colp = w0t + (n << 6) + 16 * halfsel;
        v16h bf0 = *(const v16h*)(colp);        // K-step 0
        v16h bf1 = *(const v16h*)(colp + 32);   // K-step 1

        v8f acc = {0.f, 0.f, 0.f, 0.f, 0.f, 0.f, 0.f, 0.f};
        acc = __builtin_amdgcn_wmma_f32_16x16x32_f16(false, a[0], false, bf0,
                                                     (short)0, acc, false, false);
        acc = __builtin_amdgcn_wmma_f32_16x16x32_f16(false, a[1], false, bf1,
                                                     (short)0, acc, false, false);

        const float bias = b0[type * NNEU + n];
        const float w1v  = W1[type * NNEU + n];
#pragma unroll
        for (int r = 0; r < 8; ++r)
            e += mask[r] * (fast_tanh(acc[r] + bias) * w1v);
    }

    // wave32 reduction
#pragma unroll
    for (int s = 16; s > 0; s >>= 1) e += __shfl_down(e, s, 32);
    if (lane == 0) partials[tile] = e;
}

__global__ __launch_bounds__(256) void tnep_reduce(const float* __restrict__ partials,
                                                   const int* __restrict__ tileCountPtr,
                                                   const float* __restrict__ b1,
                                                   float* __restrict__ out, int nAtoms) {
    __shared__ float sm[256];
    const int nT = *tileCountPtr;
    float s = 0.0f;
    for (int i = threadIdx.x; i < nT; i += 256) s += partials[i];
    sm[threadIdx.x] = s;
    __syncthreads();
    for (int w = 128; w > 0; w >>= 1) {
        if ((int)threadIdx.x < w) sm[threadIdx.x] += sm[threadIdx.x + w];
        __syncthreads();
    }
    if (threadIdx.x == 0) out[0] = sm[0] + (float)nAtoms * b1[0];
}

// ---------------- launcher ----------------

extern "C" void kernel_launch(void* const* d_in, const int* in_sizes, int n_in,
                              void* d_out, int out_size, void* d_ws, size_t ws_size,
                              hipStream_t stream) {
    const float* desc = (const float*)d_in[0];
    const float* W0   = (const float*)d_in[1];
    const float* b0   = (const float*)d_in[2];
    const float* W1   = (const float*)d_in[3];
    const float* b1   = (const float*)d_in[4];
    const int*   Z    = (const int*)d_in[5];

    const int n        = in_sizes[0] / DIMQ;            // number of atoms
    const int maxTiles = n / 16 + NTYPES;               // padded upper bound
    const int maxPad   = maxTiles * 16;

    auto align256 = [](size_t x) { return (x + 255) & ~(size_t)255; };
    char*  ws        = (char*)d_ws;
    size_t o         = 0;
    int*   counts    = (int*)(ws + o); o += 256;
    int*   cursors   = (int*)(ws + o); o += 256;
    int*   offsets   = (int*)(ws + o); o += 256;
    int*   tileCount = (int*)(ws + o); o += 256;
    int*   tileType  = (int*)(ws + o); o += align256((size_t)maxTiles * 4);
    float* partials  = (float*)(ws + o); o += align256((size_t)maxTiles * 4);
    int*   sortedIdx = (int*)(ws + o); o += align256((size_t)maxPad * 4);
    _Float16* W0T    = (_Float16*)(ws + o);

    const int TB = 256;
    tnep_init<<<(maxPad + TB - 1) / TB, TB, 0, stream>>>(sortedIdx, counts, cursors, maxPad);
    tnep_hist<<<(n + TB - 1) / TB, TB, 0, stream>>>(Z, counts, n);
    tnep_setup<<<1, TB, 0, stream>>>(counts, offsets, tileType, tileCount);
    tnep_scatter<<<(n + TB - 1) / TB, TB, 0, stream>>>(Z, offsets, cursors, sortedIdx, n);
    tnep_wprep<<<(NTYPES * DIMQ * NNEU + TB - 1) / TB, TB, 0, stream>>>(W0, W0T);

    const int gemmThreads = maxTiles * 32;              // one wave32 per tile
    tnep_gemm<<<(gemmThreads + TB - 1) / TB, TB, 0, stream>>>(
        desc, b0, W1, W0T, sortedIdx, tileType, tileCount, partials);

    tnep_reduce<<<1, TB, 0, stream>>>(partials, tileCount, b1, (float*)d_out, n);
}
